// mdlstm_44530220925033
// MI455X (gfx1250) — matrix operand order repaired and checked by reading, not compiled, not run on hardware
//
#include <hip/hip_runtime.h>

// ---------------------------------------------------------------------------
// LSTM recurrence on gfx1250 (MI455X): gates^T = Whh * h^T  via WMMA f16,
// rank-1 (w_ih*x + bias) init via WMMA f32 16x16x4, register-resident c/h,
// half-wave shuffle instead of LDS transpose.
// ---------------------------------------------------------------------------

typedef __attribute__((ext_vector_type(16))) _Float16 v16h;
typedef __attribute__((ext_vector_type(8)))  float    v8f;
typedef __attribute__((ext_vector_type(2)))  float    v2f;
typedef __attribute__((ext_vector_type(2)))  __fp16   f16x2; // cvt_pkrtz result type

#define T_STEPS 2048
#define HID     32

__device__ __forceinline__ float fexp2e(float x) {
    return __builtin_amdgcn_exp2f(x * 1.44269504088896340736f); // e^x
}
__device__ __forceinline__ float frcp(float x) { return __builtin_amdgcn_rcpf(x); }
__device__ __forceinline__ float fsig(float x) { return frcp(1.0f + fexp2e(-x)); }
__device__ __forceinline__ float ftanh(float x) {
    // tanh(x) = 1 - 2/(e^{2x}+1); saturates correctly for large |x|
    return 1.0f - 2.0f * frcp(fexp2e(2.0f * x) + 1.0f);
}

union PK  { f16x2 h; unsigned int u; };
union BH  { v16h h; unsigned int u[8]; };

__global__ __launch_bounds__(32)
void lstm_wmma_kernel(const float* __restrict__ x,      // (B, T)
                      const float* __restrict__ w_ih,   // (128, 1)
                      const float* __restrict__ w_hh,   // (128, 32)
                      const float* __restrict__ b_ih,   // (128,)
                      const float* __restrict__ b_hh,   // (128,)
                      float* __restrict__ out)          // (B, T*32)
{
    const int lane   = threadIdx.x;   // 0..31, wave32
    const int m      = lane & 15;     // batch row within tile (D-layout N)
    const int halfId = lane >> 4;     // 0: lanes 0-15, 1: lanes 16-31
    const int base   = blockIdx.x * 16;

    // ---- loop-invariant A-operand: W_hh in f16 16x32 A-layout, 8 tiles ----
    // A 16x32 f16 (contiguous k, as in bh): lanes<16 row=M=lane, element e -> K=e
    //                     lanes>=16 same row, K offset by +16.
    v16h wa[8];
    #pragma unroll
    for (int g = 0; g < 8; ++g) {
        const int row = 16 * g + m;   // gate row q
        v16h av;
        #pragma unroll
        for (int e = 0; e < 16; ++e) {
            const int k = e + (halfId ? 16 : 0);
            av[e] = (_Float16)w_hh[row * HID + k];
        }
        wa[g] = av;
    }

#if __has_builtin(__builtin_amdgcn_wmma_f32_16x16x4_f32)
    // rank-1 init matmul: A = 16x4 f32: K0 = w_ih[q], K1 = bias[q], K2/K3 = 0
    // lanes<16 hold (K0,K1); lanes>=16 hold (K2,K3) = zeros.
    v2f a2[8];
    #pragma unroll
    for (int g = 0; g < 8; ++g) {
        const int row = 16 * g + m;
        v2f av;
        av[0] = halfId ? 0.0f : w_ih[row];
        av[1] = halfId ? 0.0f : (b_ih[row] + b_hh[row]);
        a2[g] = av;
    }
#else
    // fallback: same rank-2 trick via f16 16x16x32 WMMA (K0=w_ih, K1=bias)
    v16h a2h[8];
    #pragma unroll
    for (int g = 0; g < 8; ++g) {
        const int row = 16 * g + m;
        v16h av;
        #pragma unroll
        for (int e = 0; e < 16; ++e) av[e] = (_Float16)0.0f;
        if (!halfId) {
            av[0] = (_Float16)w_ih[row];
            av[1] = (_Float16)(b_ih[row] + b_hh[row]);
        }
        a2h[g] = av;
    }
#endif

    // ---- register-resident state ----
    float cA[8], cB[8];
    #pragma unroll
    for (int r = 0; r < 8; ++r) { cA[r] = 0.0f; cB[r] = 0.0f; }

    BH bh;                              // h^T in B-layout (32x16 f16)
    #pragma unroll
    for (int j = 0; j < 8; ++j) bh.u[j] = 0u;

    const float* xrow   = x   + (size_t)(base + m) * T_STEPS;
    float*       rowOut = out + (size_t)(base + m) * (size_t)(T_STEPS * HID);

    float xc = xrow[0];                 // software-pipelined x_t

    for (int t = 0; t < T_STEPS; ++t) {
        // prefetch next step's input (latency hidden behind WMMA/elementwise)
        const int tn = (t + 1 < T_STEPS) ? (t + 1) : (T_STEPS - 1);
        const float xn = xrow[tn];

        v8f d[8];
#if __has_builtin(__builtin_amdgcn_wmma_f32_16x16x4_f32)
        // B = 4x16 f32: VGPR0 = row K0 (x[m]) lanes<16 / K2 (0) lanes>=16
        //               VGPR1 = row K1 (1.0)  lanes<16 / K3 (0) lanes>=16
        v2f b2;
        b2[0] = halfId ? 0.0f : xc;
        b2[1] = halfId ? 0.0f : 1.0f;
        const v8f zc = {};
        #pragma unroll
        for (int g = 0; g < 8; ++g)
            d[g] = __builtin_amdgcn_wmma_f32_16x16x4_f32(
                       false, a2[g], false, b2, (short)0, zc, false, false);
#else
        v16h b2h;
        #pragma unroll
        for (int e = 0; e < 16; ++e) b2h[e] = (_Float16)0.0f;
        if (!halfId) { b2h[0] = (_Float16)xc; b2h[1] = (_Float16)1.0f; }
        const v8f zc = {};
        #pragma unroll
        for (int g = 0; g < 8; ++g)
            d[g] = __builtin_amdgcn_wmma_f32_16x16x32_f16(
                       false, a2h[g], false, b2h, (short)0, zc, false, false);
#endif
        // main recurrence matmul: gates^T += Whh * h^T
        #pragma unroll
        for (int g = 0; g < 8; ++g)
            d[g] = __builtin_amdgcn_wmma_f32_16x16x32_f16(
                       false, wa[g], false, bh.h, (short)0, d[g], false, false);

        // ---- elementwise gates: tiles 0,1=i  2,3=f  4,5=g  6,7=o ----
        // lane holds batch m; group A: hcol = r + 8*halfId; group B: +16
        float hA[8], hB[8];
        #pragma unroll
        for (int r = 0; r < 8; ++r) {
            const float iA = fsig(d[0][r]);
            const float fA = fsig(d[2][r]);
            const float gA = ftanh(d[4][r]);
            const float oA = fsig(d[6][r]);
            cA[r] = fA * cA[r] + iA * gA;
            hA[r] = oA * ftanh(cA[r]);

            const float iB = fsig(d[1][r]);
            const float fB = fsig(d[3][r]);
            const float gB = ftanh(d[5][r]);
            const float oB = fsig(d[7][r]);
            cB[r] = fB * cB[r] + iB * gB;
            hB[r] = oB * ftanh(cB[r]);
        }

        // ---- stream h_t to output: 4 x b128 per lane, contiguous hcols ----
        {
            float* op = rowOut + t * HID;
            float4 s0 = make_float4(hA[0], hA[1], hA[2], hA[3]);
            float4 s1 = make_float4(hA[4], hA[5], hA[6], hA[7]);
            float4 s2 = make_float4(hB[0], hB[1], hB[2], hB[3]);
            float4 s3 = make_float4(hB[4], hB[5], hB[6], hB[7]);
            *(float4*)(op + 8 * halfId)          = s0;
            *(float4*)(op + 8 * halfId + 4)      = s1;
            *(float4*)(op + 16 + 8 * halfId)     = s2;
            *(float4*)(op + 16 + 8 * halfId + 4) = s3;
        }

        // ---- repack h into B-layout for next step: pack f16 pairs, swap
        //      halves across lanes (l ^ 16) — no LDS round-trip ----
        unsigned int ownA[4], ownB[4];
        #pragma unroll
        for (int j = 0; j < 4; ++j) {
            PK pa; pa.h = __builtin_amdgcn_cvt_pkrtz(hA[2 * j], hA[2 * j + 1]);
            PK pb; pb.h = __builtin_amdgcn_cvt_pkrtz(hB[2 * j], hB[2 * j + 1]);
            ownA[j] = pa.u;
            ownB[j] = pb.u;
        }
        #pragma unroll
        for (int j = 0; j < 4; ++j) {
            // lanes<16 send hB (K16-23 -> partner), lanes>=16 send hA (K8-15)
            const unsigned int send = halfId ? ownA[j] : ownB[j];
            const unsigned int rec =
                (unsigned int)__shfl_xor((int)send, 16, 32);
            bh.u[j]     = halfId ? rec     : ownA[j]; // elements 0-7
            bh.u[4 + j] = halfId ? ownB[j] : rec;     // elements 8-15
        }

        xc = xn;
    }
}

extern "C" void kernel_launch(void* const* d_in, const int* in_sizes, int n_in,
                              void* d_out, int out_size, void* d_ws, size_t ws_size,
                              hipStream_t stream) {
    const float* x    = (const float*)d_in[0];   // (B, T)
    const float* wih  = (const float*)d_in[1];   // (128, 1)
    const float* whh  = (const float*)d_in[2];   // (128, 32)
    const float* bih  = (const float*)d_in[3];   // (128,)
    const float* bhh  = (const float*)d_in[4];   // (128,)
    float*       out  = (float*)d_out;           // (B, T*32)

    const int Bn = in_sizes[0] / T_STEPS;        // 2048
    const int tiles = Bn / 16;                   // 128 waves, one per block

    lstm_wmma_kernel<<<dim3(tiles), dim3(32), 0, stream>>>(
        x, wih, whh, bih, bhh, out);
}
